// GATmodel_50517405336008
// MI455X (gfx1250) — matrix operand order repaired and checked by reading, not compiled, not run on hardware
//
#include <hip/hip_runtime.h>
#include <hip/hip_bf16.h>

// ---------------------------------------------------------------------------
// GAT forward for MI455X (gfx1250, wave32, WMMA f32_16x16x32_f16 + TDM)
// B=32, N=512, F=300(->320), H=8, D=64, HID=512, OUT=256
//
// All GEMM operands pre-swizzled into WMMA fragments with contiguous k order (k = 16*(lane/16) + elem) in
// f16: each A/B fragment is one per-lane 32B load (2x global_load_b128).
// Attention softmax fused (scores never touch memory). Support tile staged
// into LDS by the Tensor Data Mover (tensor_load_to_lds + s_wait_tensorcnt),
// shared by all head-waves. Working set (~100 MB) fits in 192 MB L2.
// ---------------------------------------------------------------------------

#define Bk 32
#define Nk 512
#define Fk 300
#define FKC 10          // ceil(300/32)
#define Hk 8
#define Dk 64
#define HIDk 512
#define OUTk 256
#define ALPHAk 0.02f

typedef __attribute__((ext_vector_type(16))) _Float16 v16h;
typedef __attribute__((ext_vector_type(8)))  _Float16 v8h;
typedef __attribute__((ext_vector_type(8)))  float    v8f;

__device__ __forceinline__ v8f wmma_f16(v16h a, v16h b, v8f c) {
  return __builtin_amdgcn_wmma_f32_16x16x32_f16(false, a, false, b, (short)0, c,
                                                false, false);
}

__device__ __forceinline__ float lrelu(float v) {
  return v > 0.f ? v : ALPHAk * v;
}

__device__ __forceinline__ void atomicMaxFloat(float* addr, float v) {
  if (v >= 0.f) atomicMax((int*)addr, __float_as_int(v));
  else          atomicMin((unsigned int*)addr, __float_as_uint(v));
}

// ---------------------------------------------------------------------------
// Tensor Data Mover: DMA a 2D f32 tile (rows x 512, row-contiguous) into LDS.
// D# built per ISA 08_async_tensor §8. Issued by one wave; completion via
// TENSORcnt. workgroup_mask=0 (not in a cluster).
// ---------------------------------------------------------------------------
#if defined(__has_builtin)
#if __has_builtin(__builtin_amdgcn_tensor_load_to_lds)
#define HAVE_TDM 1
#endif
#endif

#ifdef HAVE_TDM
typedef __attribute__((ext_vector_type(4))) unsigned int u32x4;
typedef __attribute__((ext_vector_type(8))) int i32x8;
typedef __attribute__((ext_vector_type(4))) int i32x4;

__device__ __forceinline__ void tdm_load_rows512_f32(const float* gptr,
                                                     void* lds, int rows) {
  const unsigned long long ga = (unsigned long long)(uintptr_t)gptr;
  const unsigned lds_addr = (unsigned)(uintptr_t)lds;  // low 32 bits = LDS off
  u32x4 g0;
  g0[0] = 1u;                                   // count=1, user descriptor
  g0[1] = lds_addr;                             // D#[63:32]  lds_addr
  g0[2] = (unsigned)(ga & 0xFFFFFFFFu);         // D#[95:64]  global_addr lo
  g0[3] = (unsigned)((ga >> 32) & 0x01FFFFFFu)  // D#[120:96] global_addr hi
          | (2u << 30);                         // D#[127:126] type=2 (image)
  i32x8 g1;
  g1[0] = 0x00020000;                           // data_size=2 (4B); mask=0
  g1[1] = (int)(512u << 16);                    // tensor_dim0 = 512
  g1[2] = (int)(512u << 16);                    // tensor_dim1 = 512
  g1[3] = (int)(512u << 16);                    // tile_dim0   = 512
  g1[4] = rows;                                 // tile_dim1   = rows
  g1[5] = 512;                                  // tensor_dim0_stride = 512
  g1[6] = 0;
  g1[7] = 0;
  const i32x4 gz = {0, 0, 0, 0};                // 2D tensor: groups 2/3 unused
#if __clang_major__ >= 23
  const i32x8 gz8 = {0, 0, 0, 0, 0, 0, 0, 0};
  __builtin_amdgcn_tensor_load_to_lds(g0, g1, gz, gz, gz8, 0);
#else
  __builtin_amdgcn_tensor_load_to_lds(g0, g1, gz, gz, 0);
#endif
}
#endif  // HAVE_TDM

// ---------------------------------------------------------------------------
// Pre-pass relayout kernels (one half per thread, output writes coalesced).
// ---------------------------------------------------------------------------
// x [B,N,F] f32 -> xA f16 A-frag layout [b][rt:32][kc:10][lane*16+half]
__global__ void p_x(const float* __restrict__ x, _Float16* __restrict__ xA) {
  const int idx = blockIdx.x * 256 + threadIdx.x;     // 5,242,880 total
  const int within = idx & 511;
  const int tile = idx >> 9;
  const int kc = tile % FKC;
  const int tmp = tile / FKC;
  const int rt = tmp & 31;
  const int b = tmp >> 5;
  const int lane = within >> 4, half = within & 15;
  const int row = rt * 16 + (lane & 15);
  const int kb = (lane >> 4) * 16;
  const int kk = kb + half;
  const int k = kc * 32 + kk;
  const float v = (k < Fk) ? x[((size_t)b * Nk + row) * Fk + k] : 0.f;
  xA[idx] = (_Float16)v;
}

// W1 [H,F,D] f32 -> w1B f16 B-frag layout [h][kc:10][nt:4][lane*16+half]
__global__ void p_w1(const float* __restrict__ W1, _Float16* __restrict__ w1B) {
  const int idx = blockIdx.x * 256 + threadIdx.x;     // 163,840 total
  const int within = idx & 511;
  const int tile = idx >> 9;
  const int nt = tile & 3;
  const int t2 = tile >> 2;
  const int kc = t2 % FKC;
  const int h = t2 / FKC;
  const int lane = within >> 4, half = within & 15;
  const int n = nt * 16 + (lane & 15);
  const int k = kc * 32 + (lane >> 4) * 16 + half;
  const float v = (k < Fk) ? W1[((size_t)h * Fk + k) * Dk + n] : 0.f;
  w1B[idx] = (_Float16)v;
}

// W2 [HID,OUT] f32 -> w2B f16 B-frag layout [kc:16][nt:16][lane*16+half]
__global__ void p_w2(const float* __restrict__ W2, _Float16* __restrict__ w2B) {
  const int idx = blockIdx.x * 256 + threadIdx.x;     // 131,072 total
  const int within = idx & 511;
  const int tile = idx >> 9;
  const int nt = tile & 15;
  const int kc = tile >> 4;
  const int lane = within >> 4, half = within & 15;
  const int n = nt * 16 + (lane & 15);
  const int k = kc * 32 + (lane >> 4) * 16 + half;
  w2B[idx] = (_Float16)W2[(size_t)k * OUTk + n];
}

// ---------------------------------------------------------------------------
// K1: Wh[b,h] = x[b] @ W1[h]; store B-frag layout for K2; fold f1/f2.
// Grid: B*H*(N/64) = 2048 blocks x 128 threads.
// ---------------------------------------------------------------------------
__global__ void __launch_bounds__(128)
k1_wh(const _Float16* __restrict__ xA, const _Float16* __restrict__ w1B,
      const float* __restrict__ a1s, const float* __restrict__ a1d,
      _Float16* __restrict__ whB, float* __restrict__ f1,
      float* __restrict__ f2) {
  const int blk = blockIdx.x;
  const int nt = blk & 7;
  const int h = (blk >> 3) & 7;
  const int b = blk >> 6;
  const int wave = threadIdx.x >> 5;
  const int lane = threadIdx.x & 31;
  const int mrow = lane & 15;
  const int hi = lane >> 4;
  const int i0 = nt * 64 + wave * 16;
  const int rt = i0 >> 4;

  v8f z = {};
  v8f acc[4] = {z, z, z, z};

  const _Float16* xa = xA + (((size_t)b * 32 + rt) * FKC) * 512 + (size_t)lane * 16;
  const _Float16* wb = w1B + ((size_t)h * FKC * 4) * 512 + (size_t)lane * 16;

  for (int kc = 0; kc < FKC; ++kc) {
    const v16h A = *(const v16h*)(xa + (size_t)kc * 512);
#pragma unroll
    for (int t = 0; t < 4; ++t) {
      const v16h Bf = *(const v16h*)(wb + ((size_t)kc * 4 + t) * 512);
      acc[t] = wmma_f16(A, Bf, acc[t]);
    }
  }

  // epilogue: packed v8h stores into whB + f1/f2 reductions
  const int kcr = i0 >> 5;
  const int laneB = ((i0 >> 4) & 1) * 16 + mrow;
  _Float16* wout =
      whB + ((((size_t)b * Hk + h) * 16 + kcr) * 4) * 512 + (size_t)laneB * 16 + hi * 8;
  const float* as = a1s + h * Dk;
  const float* ad = a1d + h * Dk;
  float s1[8], s2[8];
#pragma unroll
  for (int r = 0; r < 8; ++r) { s1[r] = 0.f; s2[r] = 0.f; }
#pragma unroll
  for (int t = 0; t < 4; ++t) {
    v8h pk;
    const int col = t * 16 + mrow;
#pragma unroll
    for (int r = 0; r < 8; ++r) {
      const float v = acc[t][r];
      pk[r] = (_Float16)v;
      s1[r] += v * as[col];
      s2[r] += v * ad[col];
    }
    *(v8h*)(wout + (size_t)t * 512) = pk;
  }
#pragma unroll
  for (int mm = 1; mm <= 8; mm <<= 1) {
#pragma unroll
    for (int r = 0; r < 8; ++r) {
      s1[r] += __shfl_xor(s1[r], mm, 32);
      s2[r] += __shfl_xor(s2[r], mm, 32);
    }
  }
  if (mrow == 0) {
#pragma unroll
    for (int r = 0; r < 8; ++r) {
      const int row = i0 + r + hi * 8;
      f1[((size_t)b * Hk + h) * Nk + row] = s1[r];
      f2[((size_t)b * Hk + h) * Nk + row] = s2[r];
    }
  }
}

// ---------------------------------------------------------------------------
// K2: per-head fused masked-softmax attention + ELU; h stored in A-frag
// layout via LDS transpose bounce. Grid: B*(N/16)=1024 blocks x 256 threads.
// Support tile DMA'd into LDS by the TDM (wave 0), overlapped with f2 stage.
// ---------------------------------------------------------------------------
__global__ void __launch_bounds__(256)
k2_attn(const float* __restrict__ support, const _Float16* __restrict__ whB,
        const float* __restrict__ f1g, const float* __restrict__ f2g,
        _Float16* __restrict__ hA) {
  __shared__ float Ssup[16 * 512];      // 32 KB
  __shared__ float Sf2[8 * 512];        // 16 KB
  __shared__ _Float16 Ht[16 * 512];     // 16 KB transpose bounce
  const int blk = blockIdx.x;
  const int tile = blk & 31;
  const int b = blk >> 5;
  const int i0 = tile * 16;
  const int tid = threadIdx.x;
  const int h = tid >> 5;
  const int lane = tid & 31;
  const int mrow = lane & 15;
  const int hi = lane >> 4;

  const float* sup = support + ((size_t)b * Nk + i0) * Nk;
#ifdef HAVE_TDM
  if (h == 0) {                       // wave 0 issues the tile DMA
    tdm_load_rows512_f32(sup, (void*)Ssup, 16);
  }
#else
  for (int i = tid; i < 16 * 512; i += 256) Ssup[i] = sup[i];
#endif
  const float* f2b = f2g + (size_t)b * Hk * Nk;
  for (int i = tid; i < 8 * 512; i += 256) Sf2[i] = f2b[i];
#ifdef HAVE_TDM
  if (h == 0) __builtin_amdgcn_s_wait_tensorcnt(0);
#endif
  __syncthreads();

  const float f1r = f1g[((size_t)b * Hk + h) * Nk + i0 + mrow];
  const float4* s4 = (const float4*)(Ssup + mrow * 512);
  const float4* f4 = (const float4*)(Sf2 + h * 512);

  // pass 1: online softmax stats (lane pair covers 256 cols each, float4)
  float m = -1e30f, s = 0.f;
  for (int c4 = hi * 64; c4 < hi * 64 + 64; ++c4) {
    const float4 sv = s4[c4];
    const float4 fv = f4[c4];
    const float ss[4] = {sv.x, sv.y, sv.z, sv.w};
    const float ff[4] = {fv.x, fv.y, fv.z, fv.w};
#pragma unroll
    for (int u = 0; u < 4; ++u) {
      if (ss[u] > 0.f) {
        const float e = lrelu(f1r + ff[u]);
        if (e > m) { s *= __expf(m - e); m = e; }
        s += __expf(e - m);
      }
    }
  }
  const float m2 = __shfl_xor(m, 16, 32);
  const float s2 = __shfl_xor(s, 16, 32);
  const float M = fmaxf(m, m2);
  const float S = s * __expf(m - M) + s2 * __expf(m2 - M);
  const float inv = 1.f / S;

  // pass 2: P @ Wh with P built per K-chunk in registers
  v8f z = {};
  v8f acc[4] = {z, z, z, z};
  const _Float16* wbb =
      whB + (((size_t)b * Hk + h) * 16 * 4) * 512 + (size_t)lane * 16;
  const int kb = hi * 16;
  for (int jc = 0; jc < 16; ++jc) {
    const int j0 = jc * 32;
    if (jc < 15) __builtin_prefetch(wbb + ((size_t)(jc + 1) * 4) * 512, 0, 0);
    const int g1i = (j0 + kb) >> 2;
    const int g2i = (j0 + 8 + kb) >> 2;
    const float4 sa0 = s4[g1i], sa1 = s4[g1i + 1];
    const float4 fa0 = f4[g1i], fa1 = f4[g1i + 1];
    const float4 sb0 = s4[g2i], sb1 = s4[g2i + 1];
    const float4 fb0 = f4[g2i], fb1 = f4[g2i + 1];
    const float se[8] = {sa0.x, sa0.y, sa0.z, sa0.w, sa1.x, sa1.y, sa1.z, sa1.w};
    const float fe[8] = {fa0.x, fa0.y, fa0.z, fa0.w, fa1.x, fa1.y, fa1.z, fa1.w};
    const float sf[8] = {sb0.x, sb0.y, sb0.z, sb0.w, sb1.x, sb1.y, sb1.z, sb1.w};
    const float fg[8] = {fb0.x, fb0.y, fb0.z, fb0.w, fb1.x, fb1.y, fb1.z, fb1.w};
    v16h A;
#pragma unroll
    for (int i = 0; i < 8; ++i) {
      const float p1 = se[i] > 0.f ? __expf(lrelu(f1r + fe[i]) - M) * inv : 0.f;
      const float p2 = sf[i] > 0.f ? __expf(lrelu(f1r + fg[i]) - M) * inv : 0.f;
      A[i] = (_Float16)p1;
      A[i + 8] = (_Float16)p2;
    }
#pragma unroll
    for (int t = 0; t < 4; ++t) {
      const v16h Bf = *(const v16h*)(wbb + ((size_t)jc * 4 + t) * 512);
      acc[t] = wmma_f16(A, Bf, acc[t]);
    }
  }

  // ELU -> LDS tile [16 rows][512 cols] (block covers all 8 heads)
#pragma unroll
  for (int t = 0; t < 4; ++t) {
#pragma unroll
    for (int r = 0; r < 8; ++r) {
      float v = acc[t][r];
      v = v > 0.f ? v : (__expf(v) - 1.f);
      Ht[(r + hi * 8) * 512 + (h * Dk + t * 16 + mrow)] = (_Float16)v;
    }
  }
  __syncthreads();

  // bounce: emit hA in A-frag layout, coalesced 2x16B stores per fragment
  for (int f = tid; f < 512; f += 256) {       // 16 kc x 32 lanes
    const int kc = f >> 5, ln = f & 31;
    const int mr = ln & 15, kbb = (ln >> 4) * 16;
    const _Float16* src = Ht + mr * 512 + kc * 32 + kbb;
    const v8h lo = *(const v8h*)(src);
    const v8h hh = *(const v8h*)(src + 8);
    _Float16* dst =
        hA + (((size_t)b * 32 + tile) * 16 + kc) * 512 + (size_t)ln * 16;
    *(v8h*)dst = lo;
    *(v8h*)(dst + 8) = hh;
  }
}

// ---------------------------------------------------------------------------
// K3: Wh2 = h @ W2; store B-frag layout for K4; fold g1/g2 via LDS atomics.
// Grid: B*(N/32)=512 blocks x 256 threads (2 row-tiles x 4 col-tiles).
// ---------------------------------------------------------------------------
__global__ void __launch_bounds__(256)
k3_proj(const _Float16* __restrict__ hA, const _Float16* __restrict__ w2B,
        const float* __restrict__ a2s, const float* __restrict__ a2d,
        _Float16* __restrict__ wh2B, float* __restrict__ g1,
        float* __restrict__ g2) {
  __shared__ float Lg1[32], Lg2[32];
  const int blk = blockIdx.x;
  const int tile = blk & 15;
  const int b = blk >> 4;
  const int tid = threadIdx.x;
  const int wave = tid >> 5;
  const int lane = tid & 31;
  const int rtw = wave & 1, ct = wave >> 1;
  const int i0 = tile * 32 + rtw * 16;
  const int n0 = ct * 64;
  const int mrow = lane & 15;
  const int hi = lane >> 4;

  if (tid < 32) { Lg1[tid] = 0.f; Lg2[tid] = 0.f; }
  __syncthreads();

  v8f z = {};
  v8f acc[4] = {z, z, z, z};
  const _Float16* ha =
      hA + (((size_t)b * 32 + (i0 >> 4)) * 16) * 512 + (size_t)lane * 16;
  const _Float16* wb = w2B + (size_t)lane * 16;
  for (int kc = 0; kc < 16; ++kc) {
    const v16h A = *(const v16h*)(ha + (size_t)kc * 512);
#pragma unroll
    for (int t = 0; t < 4; ++t) {
      const v16h Bf =
          *(const v16h*)(wb + ((size_t)kc * 16 + (n0 >> 4) + t) * 512);
      acc[t] = wmma_f16(A, Bf, acc[t]);
    }
  }

  // epilogue: packed v8h stores into wh2B + g1/g2 partials
  const int kcr = i0 >> 5;                   // == tile
  const int laneB = rtw * 16 + mrow;
  _Float16* wout = wh2B + (((size_t)b * 16 + kcr) * 16) * 512 +
                   (size_t)laneB * 16 + hi * 8;
  float s1[8], s2v[8];
#pragma unroll
  for (int r = 0; r < 8; ++r) { s1[r] = 0.f; s2v[r] = 0.f; }
#pragma unroll
  for (int t = 0; t < 4; ++t) {
    const int col = n0 + t * 16 + mrow;
    v8h pk;
#pragma unroll
    for (int r = 0; r < 8; ++r) {
      const float v = acc[t][r];
      pk[r] = (_Float16)v;
      s1[r] += v * a2s[col];
      s2v[r] += v * a2d[col];
    }
    *(v8h*)(wout + ((size_t)(n0 >> 4) + t) * 512) = pk;
  }
#pragma unroll
  for (int mm = 1; mm <= 8; mm <<= 1) {
#pragma unroll
    for (int r = 0; r < 8; ++r) {
      s1[r] += __shfl_xor(s1[r], mm, 32);
      s2v[r] += __shfl_xor(s2v[r], mm, 32);
    }
  }
  if (mrow == 0) {
#pragma unroll
    for (int r = 0; r < 8; ++r) {
      const int rl = rtw * 16 + r + hi * 8;
      atomicAdd(&Lg1[rl], s1[r]);
      atomicAdd(&Lg2[rl], s2v[r]);
    }
  }
  __syncthreads();
  if (tid < 32) {
    const int row = tile * 32 + tid;
    g1[(size_t)b * Nk + row] = Lg1[tid];
    g2[(size_t)b * Nk + row] = Lg2[tid];
  }
}

// ---------------------------------------------------------------------------
// K4: output attention (fused softmax) + ELU + LayerNorm + mean/max pooling.
// Grid: B*(N/16)=1024 blocks x 128 threads (4 waves x 64-col tiles).
// Support tile DMA'd into LDS by the TDM (wave 0).
// ---------------------------------------------------------------------------
__global__ void __launch_bounds__(128)
k4_out(const float* __restrict__ support, const _Float16* __restrict__ wh2B,
       const float* __restrict__ g1g, const float* __restrict__ g2g,
       const float* __restrict__ gamma, const float* __restrict__ beta,
       float* __restrict__ meanAcc, float* __restrict__ maxAcc) {
  __shared__ float Ssup[16 * 512];     // 32 KB
  __shared__ float Sg2[512];
  __shared__ float Lm[16], Ls[16], Lsum[16], Lsq[16];
  const int blk = blockIdx.x;
  const int tile = blk & 31;
  const int b = blk >> 5;
  const int i0 = tile * 16;
  const int tid = threadIdx.x;
  const int wave = tid >> 5;
  const int lane = tid & 31;
  const int mrow = lane & 15;
  const int hi = lane >> 4;

  const float* sup = support + ((size_t)b * Nk + i0) * Nk;
#ifdef HAVE_TDM
  if (wave == 0) {
    tdm_load_rows512_f32(sup, (void*)Ssup, 16);
  }
#else
  for (int i = tid; i < 16 * 512; i += 128) Ssup[i] = sup[i];
#endif
  const float* g2b = g2g + (size_t)b * Nk;
  for (int i = tid; i < 512; i += 128) Sg2[i] = g2b[i];
  if (tid < 16) { Lsum[tid] = 0.f; Lsq[tid] = 0.f; }
#ifdef HAVE_TDM
  if (wave == 0) __builtin_amdgcn_s_wait_tensorcnt(0);
#endif
  __syncthreads();

  const float g1r = g1g[(size_t)b * Nk + i0 + mrow];
  const float4* s4 = (const float4*)(Ssup + mrow * 512);
  const float4* f4 = (const float4*)Sg2;

  if (wave == 0) {
    float m = -1e30f, s = 0.f;
    for (int c4 = hi * 64; c4 < hi * 64 + 64; ++c4) {
      const float4 sv = s4[c4];
      const float4 fv = f4[c4];
      const float ss[4] = {sv.x, sv.y, sv.z, sv.w};
      const float ff[4] = {fv.x, fv.y, fv.z, fv.w};
#pragma unroll
      for (int u = 0; u < 4; ++u) {
        if (ss[u] > 0.f) {
          const float e = lrelu(g1r + ff[u]);
          if (e > m) { s *= __expf(m - e); m = e; }
          s += __expf(e - m);
        }
      }
    }
    const float m2 = __shfl_xor(m, 16, 32);
    const float s2 = __shfl_xor(s, 16, 32);
    const float M = fmaxf(m, m2);
    const float S = s * __expf(m - M) + s2 * __expf(m2 - M);
    if (hi == 0) { Lm[mrow] = M; Ls[mrow] = S; }
  }
  __syncthreads();

  const float M = Lm[mrow];
  const float inv = 1.f / Ls[mrow];
  const int n0 = wave * 64;

  v8f z = {};
  v8f acc[4] = {z, z, z, z};
  const _Float16* wbb = wh2B + ((size_t)b * 16 * 16) * 512 + (size_t)lane * 16;
  const int kb = hi * 16;
  for (int jc = 0; jc < 16; ++jc) {
    const int j0 = jc * 32;
    if (jc < 15)
      __builtin_prefetch(wbb + ((size_t)(jc + 1) * 16 + wave * 4) * 512, 0, 0);
    const int g1i = (j0 + kb) >> 2;
    const int g2i = (j0 + 8 + kb) >> 2;
    const float4 sa0 = s4[g1i], sa1 = s4[g1i + 1];
    const float4 fa0 = f4[g1i], fa1 = f4[g1i + 1];
    const float4 sb0 = s4[g2i], sb1 = s4[g2i + 1];
    const float4 fb0 = f4[g2i], fb1 = f4[g2i + 1];
    const float se[8] = {sa0.x, sa0.y, sa0.z, sa0.w, sa1.x, sa1.y, sa1.z, sa1.w};
    const float fe[8] = {fa0.x, fa0.y, fa0.z, fa0.w, fa1.x, fa1.y, fa1.z, fa1.w};
    const float sf[8] = {sb0.x, sb0.y, sb0.z, sb0.w, sb1.x, sb1.y, sb1.z, sb1.w};
    const float fg[8] = {fb0.x, fb0.y, fb0.z, fb0.w, fb1.x, fb1.y, fb1.z, fb1.w};
    v16h A;
#pragma unroll
    for (int i = 0; i < 8; ++i) {
      const float p1 = se[i] > 0.f ? __expf(lrelu(g1r + fe[i]) - M) * inv : 0.f;
      const float p2 = sf[i] > 0.f ? __expf(lrelu(g1r + fg[i]) - M) * inv : 0.f;
      A[i] = (_Float16)p1;
      A[i + 8] = (_Float16)p2;
    }
#pragma unroll
    for (int t = 0; t < 4; ++t) {
      const v16h Bf =
          *(const v16h*)(wbb + ((size_t)jc * 16 + wave * 4 + t) * 512);
      acc[t] = wmma_f16(A, Bf, acc[t]);
    }
  }

  // ELU + LayerNorm partials
  float sum[8], sq[8];
#pragma unroll
  for (int r = 0; r < 8; ++r) { sum[r] = 0.f; sq[r] = 0.f; }
#pragma unroll
  for (int t = 0; t < 4; ++t) {
#pragma unroll
    for (int r = 0; r < 8; ++r) {
      float v = acc[t][r];
      v = v > 0.f ? v : (__expf(v) - 1.f);
      acc[t][r] = v;
      sum[r] += v;
      sq[r] += v * v;
    }
  }
#pragma unroll
  for (int mm = 1; mm <= 8; mm <<= 1) {
#pragma unroll
    for (int r = 0; r < 8; ++r) {
      sum[r] += __shfl_xor(sum[r], mm, 32);
      sq[r] += __shfl_xor(sq[r], mm, 32);
    }
  }
  if (mrow == 0) {
#pragma unroll
    for (int r = 0; r < 8; ++r) {
      const int rl = r + hi * 8;
      atomicAdd(&Lsum[rl], sum[r]);
      atomicAdd(&Lsq[rl], sq[r]);
    }
  }
  __syncthreads();

#pragma unroll
  for (int t = 0; t < 4; ++t) {
#pragma unroll
    for (int r = 0; r < 8; ++r) {
      const int rl = r + hi * 8;
      const float mu = Lsum[rl] * (1.f / (float)OUTk);
      const float var = Lsq[rl] * (1.f / (float)OUTk) - mu * mu;
      const int col = n0 + t * 16 + mrow;
      const float y =
          (acc[t][r] - mu) * rsqrtf(var + 1e-5f) * gamma[col] + beta[col];
      atomicAdd(&meanAcc[(size_t)b * OUTk + col], y * (1.f / (float)Nk));
      atomicMaxFloat(&maxAcc[(size_t)b * OUTk + col], y);
    }
  }
}

// ---------------------------------------------------------------------------
__global__ void k0_init(float* meanAcc, float* maxAcc) {
  const int i = blockIdx.x * blockDim.x + threadIdx.x;
  if (i < Bk * OUTk) { meanAcc[i] = 0.f; maxAcc[i] = -3.0e38f; }
}

__global__ void k5_final(const float* meanAcc, const float* maxAcc,
                         float* out) {
  const int i = blockIdx.x * blockDim.x + threadIdx.x;
  if (i < Bk * OUTk) out[i] = meanAcc[i] + maxAcc[i];
}

// ---------------------------------------------------------------------------
extern "C" void kernel_launch(void* const* d_in, const int* in_sizes, int n_in,
                              void* d_out, int out_size, void* d_ws,
                              size_t ws_size, hipStream_t stream) {
  (void)in_sizes; (void)n_in; (void)out_size; (void)ws_size;
  const float* x     = (const float*)d_in[0];
  const float* sup   = (const float*)d_in[1];
  // d_in[2] = mask, unused by the reference math
  const float* W1    = (const float*)d_in[3];
  const float* a1s   = (const float*)d_in[4];
  const float* a1d   = (const float*)d_in[5];
  const float* W2    = (const float*)d_in[6];
  const float* a2s   = (const float*)d_in[7];
  const float* a2d   = (const float*)d_in[8];
  const float* gamma = (const float*)d_in[9];
  const float* beta  = (const float*)d_in[10];
  float* out = (float*)d_out;

  char* ws = (char*)d_ws;
  _Float16* xA   = (_Float16*)ws; ws += (size_t)Bk * 32 * FKC * 512 * 2;  // 10.5 MB
  _Float16* w1B  = (_Float16*)ws; ws += (size_t)Hk * FKC * 4 * 512 * 2;   // 0.33 MB
  _Float16* w2B  = (_Float16*)ws; ws += (size_t)16 * 16 * 512 * 2;        // 0.26 MB
  _Float16* whB  = (_Float16*)ws; ws += (size_t)Bk * Hk * 16 * 4 * 512 * 2;  // 16 MB
  _Float16* hA   = (_Float16*)ws; ws += (size_t)Bk * 32 * 16 * 512 * 2;      // 16 MB
  _Float16* wh2B = (_Float16*)ws; ws += (size_t)Bk * 16 * 16 * 512 * 2;      //  8 MB
  float* f1      = (float*)ws;    ws += (size_t)Bk * Hk * Nk * 4;
  float* f2      = (float*)ws;    ws += (size_t)Bk * Hk * Nk * 4;
  float* g1      = (float*)ws;    ws += (size_t)Bk * Nk * 4;
  float* g2      = (float*)ws;    ws += (size_t)Bk * Nk * 4;
  float* meanAcc = (float*)ws;    ws += (size_t)Bk * OUTk * 4;
  float* maxAcc  = (float*)ws;    ws += (size_t)Bk * OUTk * 4;

  p_x <<<(Bk * 32 * FKC * 512) / 256, 256, 0, stream>>>(x, xA);
  p_w1<<<(Hk * FKC * 4 * 512) / 256, 256, 0, stream>>>(W1, w1B);
  p_w2<<<(16 * 16 * 512) / 256, 256, 0, stream>>>(W2, w2B);
  k0_init<<<32, 256, 0, stream>>>(meanAcc, maxAcc);
  k1_wh<<<Bk * Hk * (Nk / 64), 128, 0, stream>>>(xA, w1B, a1s, a1d, whB, f1, f2);
  k2_attn<<<Bk * (Nk / 16), 256, 0, stream>>>(sup, whB, f1, f2, hA);
  k3_proj<<<Bk * (Nk / 32), 256, 0, stream>>>(hA, w2B, a2s, a2d, wh2B, g1, g2);
  k4_out<<<Bk * (Nk / 16), 128, 0, stream>>>(sup, wh2B, g1, g2, gamma, beta,
                                             meanAcc, maxAcc);
  k5_final<<<32, 256, 0, stream>>>(meanAcc, maxAcc, out);
}
